// Attention_1726576856766
// MI455X (gfx1250) — compile-verified, matrix operand order checked by reading, not run on hardware
//
#include <hip/hip_runtime.h>
#include <hip/hip_bf16.h>
#include <math.h>

typedef _Float16 half_t;
typedef __attribute__((ext_vector_type(16))) _Float16 v16h;
typedef __attribute__((ext_vector_type(8)))  float    v8f;

#define HEADS_C 8
#define DH_C    32
#define B_C     2
#define N_C     2048
#define DIM_C   256
#define INNER_C 256
#define BH_C    (B_C * HEADS_C)   // 16
#define NEG_BIG (-3.0e38f)

// Both the f16 A(16x32) and B(32x16) WMMA layouts reduce to: lane holds two
// contiguous 8-half chunks of the 32-long dimension, at offsets (lane/16)*8
// and (lane/16)*8 + 16.  One loader serves A tiles, B tiles, and LDS P tiles.
static __device__ __forceinline__ v16h load_chunks16(const half_t* base, int lane) {
  const int koff = (lane >> 4) << 3;
  union U { uint4 u[2]; v16h v; } r;
  r.u[0] = *(const uint4*)(base + koff);
  r.u[1] = *(const uint4*)(base + koff + 16);
  return r.v;
}

__global__ void cvt_f32_to_f16_kernel(const float* __restrict__ src,
                                      half_t* __restrict__ dst, int n) {
  int i = blockIdx.x * blockDim.x + threadIdx.x;
  if (i < n) dst[i] = (half_t)src[i];
}

// dst[n*K + k] = (half) src[k*N + n]   (transpose + convert, K rows, Nc cols)
__global__ void transpose_f32_to_f16_kernel(const float* __restrict__ src,
                                            half_t* __restrict__ dst,
                                            int K, int Nc) {
  int i = blockIdx.x * blockDim.x + threadIdx.x;
  if (i < K * Nc) {
    int k = i / Nc, n = i % Nc;
    dst[n * K + k] = (half_t)src[i];
  }
}

// ---------------------------------------------------------------------------
// Projection GEMM: xh[4096x256] (f16) x WTproj (1024 concat cols, transposed)
// -> q (scaled, f16 [b][h][n][dh]), k (f16 [b][h][n][dh]),
//    vT (f16 [b][h][dh][n]), gates (f32 [b*n][256], sigmoid applied)
// ---------------------------------------------------------------------------
__global__ __launch_bounds__(256) void proj_kernel(
    const half_t* __restrict__ xh, const half_t* __restrict__ WTproj,
    const float* __restrict__ bg,
    half_t* __restrict__ qbuf, half_t* __restrict__ kbuf,
    half_t* __restrict__ vT, float* __restrict__ gates) {
  const int lane = threadIdx.x & 31;
  const int wave = (blockIdx.x * blockDim.x + threadIdx.x) >> 5;
  const int ct = wave & 63;   // 64 col tiles over 1024 concat cols
  const int rt = wave >> 6;   // 256 row tiles over b*n = 4096 rows
  const int row = rt * 16 + (lane & 15);
  const int col = ct * 16 + (lane & 15);

  const half_t* abase = xh + (size_t)row * DIM_C;
  const half_t* bbase = WTproj + (size_t)col * DIM_C;

  v8f c = {};
#pragma unroll
  for (int kk = 0; kk < DIM_C; kk += 32) {
    v16h a = load_chunks16(abase + kk, lane);
    v16h b = load_chunks16(bbase + kk, lane);
    c = __builtin_amdgcn_wmma_f32_16x16x32_f16(false, a, false, b,
                                               (short)0, c, false, false);
  }

  const float scale = 0.17677669529663687f;  // 32^-0.5 folded into q
#pragma unroll
  for (int r = 0; r < 8; r++) {
    const int grow = rt * 16 + r + ((lane >> 4) << 3);  // global row 0..4095
    const int bidx = grow >> 11;                        // batch
    const int nrow = grow & (N_C - 1);
    const float val = c[r];
    if (col < 256) {            // Q
      int h = col >> 5, d = col & 31;
      qbuf[(((size_t)(bidx * HEADS_C + h) * N_C) + nrow) * DH_C + d] =
          (half_t)(val * scale);
    } else if (col < 512) {     // K
      int cc = col - 256; int h = cc >> 5, d = cc & 31;
      kbuf[(((size_t)(bidx * HEADS_C + h) * N_C) + nrow) * DH_C + d] = (half_t)val;
    } else if (col < 768) {     // V stored transposed [b][h][dh][n]
      int cc = col - 512; int h = cc >> 5, d = cc & 31;
      vT[((size_t)(bidx * HEADS_C + h) * DH_C + d) * N_C + nrow] = (half_t)val;
    } else {                    // gates = sigmoid(x Wg + bg)
      int cc = col - 768;
      gates[(size_t)grow * INNER_C + cc] = 1.0f / (1.0f + __expf(-(val + bg[cc])));
    }
  }
}

// ---------------------------------------------------------------------------
// Flash attention, split-K over keys.  One wave per (part, b, h, 16-row query
// tile); each wave covers N/parts keys in chunks of 32 and emits partial
// (m, s, unnormalized acc) for a later combine.  Split-K exists purely to
// raise wave count for the 268 MB bias stream (2048 -> parts*2048 waves).
// ---------------------------------------------------------------------------
__global__ __launch_bounds__(256) void attn_partial_kernel(
    const half_t* __restrict__ qbuf, const half_t* __restrict__ kbuf,
    const half_t* __restrict__ vT, const float* __restrict__ attn_bias,
    const unsigned char* __restrict__ mask,
    float* __restrict__ pacc, float* __restrict__ pm, float* __restrict__ ps,
    int keys_per_part) {
  __shared__ __align__(16) half_t Pbuf[8 * 16 * 32];  // per-wave 16x32 P tile
  const int lane  = threadIdx.x & 31;
  const int wslot = threadIdx.x >> 5;
  const int wave  = (blockIdx.x * blockDim.x + threadIdx.x) >> 5;
  const int qt   = wave & 127;        // 128 query tiles
  const int h    = (wave >> 7) & 7;
  const int b    = (wave >> 10) & 1;
  const int part = wave >> 11;
  const int bh   = b * HEADS_C + h;

  half_t* P = Pbuf + wslot * (16 * 32);
  const int qrow0 = qt * 16;
  const int j0 = part * keys_per_part;
  const int j1 = j0 + keys_per_part;

  // Q tile (A layout), loaded once, reused for all key chunks.
  const half_t* qbase =
      qbuf + ((size_t)bh * N_C + qrow0 + (lane & 15)) * DH_C;
  const v16h aq = load_chunks16(qbase, lane);

  bool mrow[8];
#pragma unroll
  for (int r = 0; r < 8; r++)
    mrow[r] = mask[b * N_C + qrow0 + r + ((lane >> 4) << 3)] != 0;

  float m[8], s[8];
  v8f acc0 = {}, acc1 = {};
#pragma unroll
  for (int r = 0; r < 8; r++) { m[r] = NEG_BIG; s[r] = 0.0f; }

  const float* biasbase = attn_bias + ((size_t)bh * N_C + qrow0) * N_C;

  for (int j = j0; j < j1; j += 32) {
    // ---- scores: Q x K^T for two 16-key tiles ----
    const half_t* kb0 = kbuf + ((size_t)bh * N_C + j + (lane & 15)) * DH_C;
    const v16h bk0 = load_chunks16(kb0, lane);
    const v16h bk1 = load_chunks16(kb0 + 16 * DH_C, lane);
    v8f z = {};
    v8f s0 = __builtin_amdgcn_wmma_f32_16x16x32_f16(false, aq, false, bk0,
                                                    (short)0, z, false, false);
    v8f s1 = __builtin_amdgcn_wmma_f32_16x16x32_f16(false, aq, false, bk1,
                                                    (short)0, z, false, false);

    const bool mj0 = mask[b * N_C + j + (lane & 15)] != 0;
    const bool mj1 = mask[b * N_C + j + 16 + (lane & 15)] != 0;

    float sc0[8], sc1[8];
#pragma unroll
    for (int r = 0; r < 8; r++) {
      const int M = r + ((lane >> 4) << 3);
      const float* brow = biasbase + (size_t)M * N_C + j;
      const float b0 = __builtin_nontemporal_load(brow + (lane & 15));
      const float b1 = __builtin_nontemporal_load(brow + 16 + (lane & 15));
      sc0[r] = (mrow[r] && mj0) ? (s0[r] + b0) : NEG_BIG;
      sc1[r] = (mrow[r] && mj1) ? (s1[r] + b1) : NEG_BIG;
    }

    // ---- online softmax: row max / rescale (rows live in 16-lane halves) ----
    float alpha[8];
#pragma unroll
    for (int r = 0; r < 8; r++) {
      float v = fmaxf(sc0[r], sc1[r]);
      v = fmaxf(v, __shfl_xor(v, 1));
      v = fmaxf(v, __shfl_xor(v, 2));
      v = fmaxf(v, __shfl_xor(v, 4));
      v = fmaxf(v, __shfl_xor(v, 8));
      const float mnew = fmaxf(m[r], v);
      alpha[r] = __expf(m[r] - mnew);
      m[r] = mnew;
    }

#pragma unroll
    for (int r = 0; r < 8; r++) {
      const int M = r + ((lane >> 4) << 3);
      const float p0 = __expf(sc0[r] - m[r]);
      const float p1 = __expf(sc1[r] - m[r]);
      P[M * 32 + (lane & 15)]      = (half_t)p0;
      P[M * 32 + 16 + (lane & 15)] = (half_t)p1;
      float psum = p0 + p1;
      psum += __shfl_xor(psum, 1);
      psum += __shfl_xor(psum, 2);
      psum += __shfl_xor(psum, 4);
      psum += __shfl_xor(psum, 8);
      s[r] = s[r] * alpha[r] + psum;
      acc0[r] = acc0[r] * alpha[r];
      acc1[r] = acc1[r] * alpha[r];
    }

    asm volatile("s_wait_dscnt 0" ::: "memory");  // P stores -> P reload

    const v16h ap = load_chunks16(P + (lane & 15) * 32, lane);
    const half_t* vb0 = vT + ((size_t)bh * DH_C + (lane & 15)) * N_C + j;
    const v16h bv0 = load_chunks16(vb0, lane);
    const v16h bv1 = load_chunks16(vb0 + 16 * N_C, lane);
    acc0 = __builtin_amdgcn_wmma_f32_16x16x32_f16(false, ap, false, bv0,
                                                  (short)0, acc0, false, false);
    acc1 = __builtin_amdgcn_wmma_f32_16x16x32_f16(false, ap, false, bv1,
                                                  (short)0, acc1, false, false);
  }

  // ---- store partials (unnormalized acc at local max m, plus m and s) ----
#pragma unroll
  for (int r = 0; r < 8; r++) {
    const int grow = qrow0 + r + ((lane >> 4) << 3);
    const size_t rowidx = ((size_t)part * BH_C + bh) * N_C + grow;
    pacc[rowidx * DH_C + (lane & 15)]      = acc0[r];
    pacc[rowidx * DH_C + 16 + (lane & 15)] = acc1[r];
    if ((lane & 15) == 0) { pm[rowidx] = m[r]; ps[rowidx] = s[r]; }
  }
}

// ---------------------------------------------------------------------------
// Combine split-K partials: one thread per (b,h,row); renormalize across
// partitions, divide by total sum, apply gates, store f16 [b*n][256].
// ---------------------------------------------------------------------------
__global__ __launch_bounds__(256) void attn_combine_kernel(
    const float* __restrict__ pacc, const float* __restrict__ pm,
    const float* __restrict__ ps, const float* __restrict__ gates,
    half_t* __restrict__ gout, int parts) {
  const int t = blockIdx.x * blockDim.x + threadIdx.x;
  if (t >= BH_C * N_C) return;
  const int bh = t / N_C, row = t % N_C;

  float mstar = NEG_BIG;
  for (int p = 0; p < parts; p++)
    mstar = fmaxf(mstar, pm[((size_t)p * BH_C + bh) * N_C + row]);

  float S = 0.0f;
  float o[DH_C];
#pragma unroll
  for (int d = 0; d < DH_C; d++) o[d] = 0.0f;

  for (int p = 0; p < parts; p++) {
    const size_t rowidx = ((size_t)p * BH_C + bh) * N_C + row;
    const float w = __expf(pm[rowidx] - mstar);
    S += w * ps[rowidx];
    const float* a = pacc + rowidx * DH_C;
#pragma unroll
    for (int d = 0; d < DH_C; d++) o[d] += w * a[d];
  }

  const float inv = 1.0f / S;
  const int b = bh >> 3, h = bh & 7;
  const size_t base = ((size_t)b * N_C + row) * INNER_C + h * DH_C;
#pragma unroll
  for (int d = 0; d < DH_C; d++)
    gout[base + d] = (half_t)(o[d] * inv * gates[base + d]);
}

// ---------------------------------------------------------------------------
// Output GEMM: gout[4096x256] (f16) x WoutT (f16, transposed) + bout -> f32
// ---------------------------------------------------------------------------
__global__ __launch_bounds__(256) void out_kernel(
    const half_t* __restrict__ gout, const half_t* __restrict__ WoutT,
    const float* __restrict__ bout, float* __restrict__ out) {
  const int lane = threadIdx.x & 31;
  const int wave = (blockIdx.x * blockDim.x + threadIdx.x) >> 5;
  const int ct = wave & 15;   // 16 col tiles over 256 cols
  const int rt = wave >> 4;   // 256 row tiles over 4096 rows

  const half_t* abase = gout + (size_t)(rt * 16 + (lane & 15)) * INNER_C;
  const half_t* bbase = WoutT + (size_t)(ct * 16 + (lane & 15)) * INNER_C;

  v8f c = {};
#pragma unroll
  for (int kk = 0; kk < INNER_C; kk += 32) {
    v16h a = load_chunks16(abase + kk, lane);
    v16h b = load_chunks16(bbase + kk, lane);
    c = __builtin_amdgcn_wmma_f32_16x16x32_f16(false, a, false, b,
                                               (short)0, c, false, false);
  }

  const int col = ct * 16 + (lane & 15);
  const float bo = bout[col];
#pragma unroll
  for (int r = 0; r < 8; r++) {
    const int grow = rt * 16 + r + ((lane >> 4) << 3);
    out[(size_t)grow * DIM_C + col] = c[r] + bo;
  }
}

extern "C" void kernel_launch(void* const* d_in, const int* in_sizes, int n_in,
                              void* d_out, int out_size, void* d_ws, size_t ws_size,
                              hipStream_t stream) {
  const float* x            = (const float*)d_in[0];
  const unsigned char* mask = (const unsigned char*)d_in[1];
  const float* attn_bias    = (const float*)d_in[2];
  const float* Wq   = (const float*)d_in[3];
  const float* Wkv  = (const float*)d_in[4];
  const float* Wg   = (const float*)d_in[5];
  const float* bg   = (const float*)d_in[6];
  const float* Wout = (const float*)d_in[7];
  const float* bout = (const float*)d_in[8];
  float* out = (float*)d_out;

  char* ws = (char*)d_ws;
  size_t off = 0;
  half_t* xh      = (half_t*)(ws + off); off += (size_t)B_C * N_C * DIM_C * 2;      // 2 MB
  half_t* WTproj  = (half_t*)(ws + off); off += (size_t)1024 * 256 * 2;             // 512 KB
  half_t* WoutT   = (half_t*)(ws + off); off += (size_t)256 * 256 * 2;              // 128 KB
  half_t* qbuf    = (half_t*)(ws + off); off += (size_t)BH_C * N_C * DH_C * 2;      // 2 MB
  half_t* kbuf    = (half_t*)(ws + off); off += (size_t)BH_C * N_C * DH_C * 2;      // 2 MB
  half_t* vTbuf   = (half_t*)(ws + off); off += (size_t)BH_C * N_C * DH_C * 2;      // 2 MB
  float*  gates   = (float*)(ws + off);  off += (size_t)B_C * N_C * INNER_C * 4;    // 4 MB
  half_t* gout    = (half_t*)(ws + off); off += (size_t)B_C * N_C * INNER_C * 2;    // 2 MB
  const size_t fixed_bytes = off;

  // Split-K partials: choose largest parts in {4,2,1} that fits ws_size.
  // ws_size is a per-harness constant, so this choice is deterministic.
  const size_t per_part =
      (size_t)BH_C * N_C * DH_C * 4 +  // pacc: 4 MB per part
      (size_t)BH_C * N_C * 4 * 2;      // pm + ps: 256 KB per part
  int parts = 4;
  while (parts > 1 && fixed_bytes + (size_t)parts * per_part > ws_size) parts >>= 1;

  float* pacc = (float*)(ws + off); off += (size_t)parts * BH_C * N_C * DH_C * 4;
  float* pm   = (float*)(ws + off); off += (size_t)parts * BH_C * N_C * 4;
  float* ps   = (float*)(ws + off); off += (size_t)parts * BH_C * N_C * 4;
  (void)in_sizes; (void)n_in; (void)out_size;

  // Prep: convert x to f16; transpose+convert all weights (contiguous B tiles).
  const int nx = B_C * N_C * DIM_C;
  cvt_f32_to_f16_kernel<<<(nx + 255) / 256, 256, 0, stream>>>(x, xh, nx);
  transpose_f32_to_f16_kernel<<<(256 * 256 + 255) / 256, 256, 0, stream>>>(Wq,  WTproj,             256, 256);
  transpose_f32_to_f16_kernel<<<(256 * 512 + 255) / 256, 256, 0, stream>>>(Wkv, WTproj + 256 * 256, 256, 512);
  transpose_f32_to_f16_kernel<<<(256 * 256 + 255) / 256, 256, 0, stream>>>(Wg,  WTproj + 768 * 256, 256, 256);
  transpose_f32_to_f16_kernel<<<(256 * 256 + 255) / 256, 256, 0, stream>>>(Wout, WoutT,             256, 256);

  // QKV + gates projection: 16384 waves = 2048 blocks of 8 waves.
  proj_kernel<<<2048, 256, 0, stream>>>(xh, WTproj, bg, qbuf, kbuf, vTbuf, gates);

  // Flash attention split-K: parts*2048 waves = parts*256 blocks of 8 waves.
  attn_partial_kernel<<<parts * 256, 256, 0, stream>>>(
      qbuf, kbuf, vTbuf, attn_bias, mask, pacc, pm, ps, N_C / parts);

  // Combine partials + gate: one thread per (b,h,row).
  attn_combine_kernel<<<(BH_C * N_C + 255) / 256, 256, 0, stream>>>(
      pacc, pm, ps, gates, gout, parts);

  // Output projection: 4096 waves = 512 blocks of 8 waves.
  out_kernel<<<512, 256, 0, stream>>>(gout, WoutT, bout, out);
}
